// MuiInt8Linear_82162724372871
// MI455X (gfx1250) — matrix operand order repaired and checked by reading, not compiled, not run on hardware
//
#include <hip/hip_runtime.h>

#define IN_F   4096
#define OUT_F  14336
#define GS     128
#define NG     (IN_F / GS)      // 32 groups per row
#define BATCH  8
#define WAVES  8                // waves per block
#define KPW    (IN_F / WAVES)   // 512 K per wave
#define GPW    (KPW / GS)       // 4 groups per wave

typedef __attribute__((ext_vector_type(16))) _Float16 v16h;
typedef __attribute__((ext_vector_type(8)))  _Float16 v8h;
typedef __attribute__((ext_vector_type(8)))  float    v8f;
typedef __attribute__((ext_vector_type(4)))  int      v4i;

// ---------------------------------------------------------------------------
// Kernel 1: RMSNorm x[8,4096] -> f16 xn[16,4096] in workspace (rows 8..15 = 0)
// ---------------------------------------------------------------------------
__global__ void __launch_bounds__(256)
rmsnorm_kernel(const float* __restrict__ x,
               const float* __restrict__ nw,
               _Float16* __restrict__ xn) {
    const int row = blockIdx.x;           // 0..15
    const int t   = threadIdx.x;          // 0..255
    if (row >= BATCH) {                   // zero-pad rows 8..15 for the M=16 tile
        for (int i = t; i < IN_F; i += 256) xn[row * IN_F + i] = (_Float16)0.0f;
        return;
    }
    __shared__ float red[256];
    float ss = 0.0f;
    for (int i = t; i < IN_F; i += 256) {
        float v = x[row * IN_F + i];
        ss += v * v;
    }
    red[t] = ss;
    __syncthreads();
    for (int s = 128; s > 0; s >>= 1) {
        if (t < s) red[t] += red[t + s];
        __syncthreads();
    }
    const float rs = rsqrtf(red[0] * (1.0f / IN_F) + 1e-6f);
    for (int i = t; i < IN_F; i += 256) {
        xn[row * IN_F + i] = (_Float16)(x[row * IN_F + i] * rs * nw[i]);
    }
}

// ---------------------------------------------------------------------------
// Kernel 2: dequant + WMMA GEMM. One block = one 16-wide N tile; 8 waves
// split K (512 each), LDS reduction, deterministic store.
// ---------------------------------------------------------------------------
__global__ void __launch_bounds__(256)
gemm_kernel(const _Float16* __restrict__ xn,     // [16, 4096] f16
            const int*      __restrict__ wq,     // [14336, 4096] int32 in 0..255
            const float*    __restrict__ smv,    // [14336*32, 2] (scale, min)
            const float*    __restrict__ bias,   // [14336]
            float*          __restrict__ out) {  // [8, 14336]
    const int lane = threadIdx.x & 31;
    const int wave = threadIdx.x >> 5;
    const int hi   = lane >> 4;          // lane half
    const int m    = lane & 15;          // A row this lane feeds
    const int o    = blockIdx.x * 16 + (lane & 15);   // output column
    const int kbeg = wave * KPW;

    const _Float16* __restrict__ arow = xn + (size_t)m * IN_F;
    const int*      __restrict__ wrow = wq + (size_t)o * IN_F;

    v8f acc = {};

    for (int g = 0; g < GPW; ++g) {
        const int gidx  = wave * GPW + g;
        const float scl = smv[((size_t)o * NG + gidx) * 2 + 0];
        const float mn  = smv[((size_t)o * NG + gidx) * 2 + 1];
        const int kg    = kbeg + g * GS;

        // prefetch next group's weight stream
        if (g + 1 < GPW)
            __builtin_prefetch(wrow + kg + GS + hi * 16 + (lane & 15) * 4, 0, 1);

#pragma unroll
        for (int c = 0; c < 4; ++c) {
            const int k0 = kg + c * 32;

            // ---- A fragment (16-bit A 16x32 layout) ----
            // lane half 0: K = k0+0..15 ; half 1: K = k0+16..31 (consecutive)
            const int c0 = k0 + hi * 16;
            v16h a;
            *(v8h*)&a       = *(const v8h*)(arow + c0);
            *((v8h*)&a + 1) = *(const v8h*)(arow + c0 + 8);

            // ---- B fragment (16-bit B 32x16 layout) ----
            // lane half 0: K = k0+0..15 ; half 1: K = k0+16..31 (consecutive)
            const int d0 = k0 + hi * 16;
            v4i q0 = __builtin_nontemporal_load((const v4i*)(wrow + d0));
            v4i q1 = __builtin_nontemporal_load((const v4i*)(wrow + d0 + 4));
            v4i q2 = __builtin_nontemporal_load((const v4i*)(wrow + d0 + 8));
            v4i q3 = __builtin_nontemporal_load((const v4i*)(wrow + d0 + 12));

            v16h b;
#pragma unroll
            for (int j = 0; j < 4; ++j) {
                b[j]      = (_Float16)fmaf((float)q0[j], scl, mn);
                b[4 + j]  = (_Float16)fmaf((float)q1[j], scl, mn);
                b[8 + j]  = (_Float16)fmaf((float)q2[j], scl, mn);
                b[12 + j] = (_Float16)fmaf((float)q3[j], scl, mn);
            }

            acc = __builtin_amdgcn_wmma_f32_16x16x32_f16(
                /*neg_a=*/false, a, /*neg_b=*/false, b,
                /*c_mod=*/(short)0, acc, /*reuse_a=*/false, /*reuse_b=*/false);
        }
    }

    // ---- deterministic cross-wave K reduction through LDS ----
    __shared__ v8f part[WAVES][32];
    part[wave][lane] = acc;
    __syncthreads();

    // C/D layout: lanes 0..15, VGPR r <-> row M=r (rows 0..7 are the real batch)
    if (wave == 0 && hi == 0) {
        v8f s = part[0][lane];
#pragma unroll
        for (int w = 1; w < WAVES; ++w) s += part[w][lane];
        const float bv = bias[o];
#pragma unroll
        for (int r = 0; r < BATCH; ++r)
            out[(size_t)r * OUT_F + o] = s[r] + bv;
    }
}

// ---------------------------------------------------------------------------
extern "C" void kernel_launch(void* const* d_in, const int* in_sizes, int n_in,
                              void* d_out, int out_size, void* d_ws, size_t ws_size,
                              hipStream_t stream) {
    (void)in_sizes; (void)n_in; (void)out_size; (void)ws_size;
    const float* x    = (const float*)d_in[0];
    const int*   wq   = (const int*)  d_in[1];
    const float* smv  = (const float*)d_in[2];
    const float* nw   = (const float*)d_in[3];
    const float* bias = (const float*)d_in[4];
    float* out = (float*)d_out;

    _Float16* xn = (_Float16*)d_ws;   // 16 * 4096 * 2B = 128 KB scratch

    rmsnorm_kernel<<<16, 256, 0, stream>>>(x, nw, xn);
    gemm_kernel<<<OUT_F / 16, 256, 0, stream>>>(xn, wq, smv, bias, out);
}
